// VanillaLSTM_3461743641071
// MI455X (gfx1250) — matrix operand order repaired and checked by reading, not compiled, not run on hardware
//
#include <hip/hip_runtime.h>

typedef __bf16 bf16_t;
typedef __attribute__((ext_vector_type(16))) __bf16 v16bf;
typedef __attribute__((ext_vector_type(8)))  __bf16 v8bf;
typedef __attribute__((ext_vector_type(8)))  float   v8f;

#define B_SZ 256
#define T_SZ 256
#define I_SZ 512
#define H_SZ 1024
#define O_SZ 5

// ---------------------------------------------------------------------------
// Async global->LDS copy of 16 bytes per lane (CDNA5 ASYNCcnt path).
// vdst = 32-bit LDS byte address (low 32 bits of generic pointer), GV mode.
// ---------------------------------------------------------------------------
__device__ __forceinline__ void async_copy16(const void* gsrc, void* lds_dst) {
    unsigned lds = (unsigned)(unsigned long long)lds_dst;
    unsigned long long ga = (unsigned long long)gsrc;
    asm volatile("global_load_async_to_lds_b128 %0, %1, off"
                 :: "v"(lds), "v"(ga)
                 : "memory");
}

__device__ __forceinline__ float fast_sigmoid(float x) {
    return 1.0f / (1.0f + __expf(-x));
}
__device__ __forceinline__ float fast_tanh(float x) {
    return 2.0f / (1.0f + __expf(-2.0f * x)) - 1.0f;   // branch-free
}

// ---------------------------------------------------------------------------
// f32 -> bf16 conversion (weights + x + initial h live bf16 in ws / L2)
// ---------------------------------------------------------------------------
__global__ void cvt_bf16_kernel(const float* __restrict__ s,
                                bf16_t* __restrict__ d, int n) {
    int i = blockIdx.x * blockDim.x + threadIdx.x;
    int stride = gridDim.x * blockDim.x;
    for (; i < n; i += stride) d[i] = (bf16_t)s[i];
}

// ---------------------------------------------------------------------------
// One LSTM timestep, fused GEMM + gates.
//   gates[g,b,h] = sum_k concat(h_prev, x_t)[b,k] * W[g,h,k] + bias
// WG tile: 64 b-rows x {4 gates x 32 h-cols}; 8 waves = (wm 0..1, gate 0..3);
// each wave: 2x2 WMMA 16x16 tiles. K loop manually unrolled by 2 so the two
// A-tile LDS buffers and the two B-fragment register sets alternate roles
// (no register rotation, no dynamic LDS indexing). A tiles arrive via async
// DMA (ASYNCcnt); B fragments are plain loads double-buffered across the
// barrier.
// ---------------------------------------------------------------------------
template <int LAYER>
__global__ __launch_bounds__(256) void lstm_step_kernel(
    const bf16_t* __restrict__ h_in_bf,   // [B,H] bf16 h_{t-1} (this layer)
    bf16_t* __restrict__ h_out_bf,        // [B,H] bf16 h_t
    float* __restrict__ h_out_f32,        // [B,H] f32 h_t (for finals)
    float* __restrict__ c_state,          // [B,H] f32 in/out
    const bf16_t* __restrict__ xsrc_bf,   // L0: x_bf [B,T,I]; L1: layer0 h_t bf16
    int t,
    const bf16_t* __restrict__ Wb,        // [4*H, K] bf16 row-major
    const float* __restrict__ bias)       // [4*H]
{
    constexpr int K  = (LAYER == 0) ? (H_SZ + I_SZ) : (2 * H_SZ);
    constexpr int NK = K / 32;            // 48 or 64, both even

    const int m_base = blockIdx.x * 64;
    const int h_base = blockIdx.y * 32;
    const int tid  = threadIdx.x;
    const int lane = tid & 31;
    const int wave = tid >> 5;
    const int wm   = wave >> 2;   // 0..1
    const int gate = wave & 3;    // 0..3

    __shared__ bf16_t Atile[2][64][40];   // double-buffered, padded rows
    __shared__ float  gbuf[4][64][32];

    v8f acc[2][2];
#pragma unroll
    for (int i = 0; i < 2; i++)
#pragma unroll
        for (int j = 0; j < 2; j++)
#pragma unroll
            for (int e = 0; e < 8; e++) acc[i][j][e] = 0.0f;

    const int srow  = tid >> 2;         // 0..63
    const int scol  = (tid & 3) * 8;    // 0,8,16,24
    const int garow = m_base + srow;
    const int half  = lane >> 4;
    const int l15   = lane & 15;

    auto src_ptr = [&](int k0) -> const bf16_t* {   // uniform region per k0
        const int kc = k0 + scol;
        if (kc < H_SZ) return h_in_bf + (size_t)garow * H_SZ + kc;
        if (LAYER == 0)
            return xsrc_bf + ((size_t)garow * T_SZ + t) * I_SZ + (kc - H_SZ);
        return xsrc_bf + (size_t)garow * H_SZ + (kc - H_SZ);
    };

    // WMMA over one staged A buffer with one B fragment set.
    auto mm = [&](const bf16_t (*At)[40], const v16bf* bfr) {
        v16bf afrag[2];
#pragma unroll
        for (int ms = 0; ms < 2; ms++) {
            const int r = wm * 32 + ms * 16 + l15;
            v8bf lo = *(const v8bf*)&At[r][half * 16];
            v8bf hi = *(const v8bf*)&At[r][half * 16 + 8];
            afrag[ms] = __builtin_shufflevector(lo, hi, 0, 1, 2, 3, 4, 5, 6, 7,
                                                8, 9, 10, 11, 12, 13, 14, 15);
        }
#pragma unroll
        for (int ms = 0; ms < 2; ms++)
#pragma unroll
            for (int ns = 0; ns < 2; ns++)
                acc[ms][ns] = __builtin_amdgcn_wmma_f32_16x16x32_bf16(
                    false, afrag[ms], false, bfr[ns], (short)0, acc[ms][ns],
                    false, false);
    };

    // --- prologue: async-stage tile 0 into buf0, prefetch B(0) ---
    async_copy16(src_ptr(0), &Atile[0][srow][scol]);
    const bf16_t* wrow0 = Wb + (size_t)(gate * H_SZ + h_base + l15) * K;
    const bf16_t* wrow1 = Wb + (size_t)(gate * H_SZ + h_base + 16 + l15) * K;
    v16bf bcur[2], bnext[2];
    bcur[0] = *(const v16bf*)(wrow0 + half * 16);
    bcur[1] = *(const v16bf*)(wrow1 + half * 16);

    for (int i = 0; i < NK; i += 2) {
        // ---- tile i (buffer 0, bcur) ----
        async_copy16(src_ptr((i + 1) * 32), &Atile[1][srow][scol]);
        bnext[0] = *(const v16bf*)(wrow0 + (i + 1) * 32 + half * 16);
        bnext[1] = *(const v16bf*)(wrow1 + (i + 1) * 32 + half * 16);
        asm volatile("s_wait_asynccnt 0x1" ::: "memory");   // tile i landed
        __syncthreads();
        mm(Atile[0], bcur);
        __syncthreads();           // buf0 reads done before reuse below

        // ---- tile i+1 (buffer 1, bnext) ----
        if (i + 2 < NK) {
            async_copy16(src_ptr((i + 2) * 32), &Atile[0][srow][scol]);
            bcur[0] = *(const v16bf*)(wrow0 + (i + 2) * 32 + half * 16);
            bcur[1] = *(const v16bf*)(wrow1 + (i + 2) * 32 + half * 16);
            asm volatile("s_wait_asynccnt 0x1" ::: "memory");
        } else {
            asm volatile("s_wait_asynccnt 0x0" ::: "memory");
        }
        __syncthreads();
        mm(Atile[1], bnext);
        __syncthreads();           // buf1 reads done before reuse next pair
    }

    // ---- scatter accumulators so all 4 gates of a cell meet in LDS ----
#pragma unroll
    for (int ms = 0; ms < 2; ms++)
#pragma unroll
        for (int ns = 0; ns < 2; ns++)
#pragma unroll
            for (int j = 0; j < 8; j++) {
                const int row = wm * 32 + ms * 16 + half * 8 + j;  // C/D layout
                const int ncl = ns * 16 + l15;
                gbuf[gate][row][ncl] = acc[ms][ns][j];
            }
    __syncthreads();

    // ---- fused gate nonlinearities + cell update (8 cells / thread) ----
#pragma unroll
    for (int i = 0; i < 8; i++) {
        const int cc  = tid + i * 256;
        const int row = cc >> 5;
        const int hc  = cc & 31;
        const int b   = m_base + row;
        const int h   = h_base + hc;
        const float gf = gbuf[0][row][hc] + bias[0 * H_SZ + h];
        const float gi = gbuf[1][row][hc] + bias[1 * H_SZ + h];
        const float gg = gbuf[2][row][hc] + bias[2 * H_SZ + h];
        const float go = gbuf[3][row][hc] + bias[3 * H_SZ + h];
        const float f  = fast_sigmoid(gf);
        const float ii = fast_sigmoid(gi);
        const float g  = fast_tanh(gg);
        const float o  = fast_sigmoid(go);
        const size_t idx = (size_t)b * H_SZ + h;
        const float cn = f * c_state[idx] + ii * g;
        const float hn = o * fast_tanh(cn);
        c_state[idx]   = cn;
        h_out_f32[idx] = hn;
        h_out_bf[idx]  = (bf16_t)hn;   // GEMM operand for the next step
    }
}

// ---------------------------------------------------------------------------
// Final projection: out[b,o] = h1_T[b,:] . W_out[o,:] + b_out[o]   (1280 dots)
// ---------------------------------------------------------------------------
__global__ void out_proj_kernel(const float* __restrict__ h,
                                const float* __restrict__ Wout,
                                const float* __restrict__ bout,
                                float* __restrict__ out) {
    const int idx = blockIdx.x * blockDim.x + threadIdx.x;
    if (idx >= B_SZ * O_SZ) return;
    const int b = idx / O_SZ, o = idx % O_SZ;
    const float* hp = h + (size_t)b * H_SZ;
    const float* wp = Wout + (size_t)o * H_SZ;
    float s = 0.0f;
    for (int k = 0; k < H_SZ; k++) s += hp[k] * wp[k];
    out[idx] = s + bout[o];
}

// ---------------------------------------------------------------------------
extern "C" void kernel_launch(void* const* d_in, const int* in_sizes, int n_in,
                              void* d_out, int out_size, void* d_ws,
                              size_t ws_size, hipStream_t stream) {
    (void)in_sizes; (void)n_in; (void)out_size; (void)ws_size;
    const float* x      = (const float*)d_in[0];
    const float* hidden = (const float*)d_in[1];
    const float* W0     = (const float*)d_in[2];
    const float* b0     = (const float*)d_in[3];
    const float* W1     = (const float*)d_in[4];
    const float* b1     = (const float*)d_in[5];
    const float* Wout   = (const float*)d_in[6];
    const float* bout   = (const float*)d_in[7];
    float* out = (float*)d_out;

    const size_t BH = (size_t)B_SZ * H_SZ;
    char* ws = (char*)d_ws;
    // State block S mirrors hidden layout [L,2,B,H]: h0 | c0 | h1 | c1 (f32).
    float* S = (float*)ws;                  ws += 4 * BH * sizeof(float);
    float* h0alt = (float*)ws;              ws += BH * sizeof(float);
    float* h1alt = (float*)ws;              ws += BH * sizeof(float);
    bf16_t* h0bf[2]; bf16_t* h1bf[2];
    h0bf[0] = (bf16_t*)ws;                  ws += BH * sizeof(bf16_t);
    h0bf[1] = (bf16_t*)ws;                  ws += BH * sizeof(bf16_t);
    h1bf[0] = (bf16_t*)ws;                  ws += BH * sizeof(bf16_t);
    h1bf[1] = (bf16_t*)ws;                  ws += BH * sizeof(bf16_t);
    bf16_t* Wb0 = (bf16_t*)ws;              ws += (size_t)4 * H_SZ * (H_SZ + I_SZ) * sizeof(bf16_t);
    bf16_t* Wb1 = (bf16_t*)ws;              ws += (size_t)4 * H_SZ * (2 * H_SZ) * sizeof(bf16_t);
    bf16_t* x_bf = (bf16_t*)ws;             // [B,T,I] bf16, 67 MB

    cvt_bf16_kernel<<<512, 256, 0, stream>>>(W0, Wb0, 4 * H_SZ * (H_SZ + I_SZ));
    cvt_bf16_kernel<<<512, 256, 0, stream>>>(W1, Wb1, 4 * H_SZ * (2 * H_SZ));
    cvt_bf16_kernel<<<4096, 256, 0, stream>>>(x, x_bf, B_SZ * T_SZ * I_SZ);
    cvt_bf16_kernel<<<256, 256, 0, stream>>>(hidden, h0bf[0], (int)BH);
    cvt_bf16_kernel<<<256, 256, 0, stream>>>(hidden + 2 * BH, h1bf[0], (int)BH);
    hipMemcpyAsync(S, hidden, 4 * BH * sizeof(float), hipMemcpyDeviceToDevice,
                   stream);

    float* c0 = S + BH;
    float* c1 = S + 3 * BH;
    float* p0f[2] = {S, h0alt};             // layer0 f32 h ping-pong (buf0 in S)
    float* p1f[2] = {S + 2 * BH, h1alt};    // layer1 f32 h ping-pong (buf0 in S)

    dim3 grid(B_SZ / 64, H_SZ / 32), block(256);
    for (int t = 0; t < T_SZ; t++) {
        const int a = t & 1, bswap = (t + 1) & 1;
        lstm_step_kernel<0><<<grid, block, 0, stream>>>(
            h0bf[a], h0bf[bswap], p0f[bswap], c0, x_bf, t, Wb0, b0);
        lstm_step_kernel<1><<<grid, block, 0, stream>>>(
            h1bf[a], h1bf[bswap], p1f[bswap], c1, h0bf[bswap], t, Wb1, b1);
    }
    // T even -> final f32 h lives in buf0 inside S: S == [h0_T|c0_T|h1_T|c1_T].
    hipMemcpyAsync(out + B_SZ * O_SZ, S, 4 * BH * sizeof(float),
                   hipMemcpyDeviceToDevice, stream);
    out_proj_kernel<<<(B_SZ * O_SZ + 255) / 256, 256, 0, stream>>>(
        S + 2 * BH, Wout, bout, out);
}
